// SparseMoELayer_75445395522222
// MI455X (gfx1250) — compile-verified, matrix operand order checked by reading, not run on hardware
//
#include <hip/hip_runtime.h>

typedef __attribute__((ext_vector_type(16))) __bf16 v16bf;
typedef __attribute__((ext_vector_type(8)))  __bf16 v8bf;
typedef __attribute__((ext_vector_type(4)))  __bf16 v4bf;
typedef __attribute__((ext_vector_type(8)))  float  v8f;
typedef __attribute__((ext_vector_type(4)))  float  v4f;

#define NUM_EXPERTS 8

// ---------------------------------------------------------------- init ------
__global__ void moe_init(float* __restrict__ out, int n, int* __restrict__ counts) {
  int i = blockIdx.x * blockDim.x + threadIdx.x;
  if (i < n) out[i] = 0.0f;
  if (i < NUM_EXPERTS) counts[i] = 0;
}

// ------------------------------------------------------------- routing ------
__global__ __launch_bounds__(256)
void moe_route(const float* __restrict__ x, const float* __restrict__ Wr,
               int* __restrict__ counts, int* __restrict__ tok_list,
               float* __restrict__ gate_list, int N, int C) {
  const int lane = threadIdx.x & 31;
  const int wave = threadIdx.x >> 5;
  const int t = blockIdx.x * (blockDim.x >> 5) + wave;
  if (t >= N) return;

  float acc[NUM_EXPERTS];
#pragma unroll
  for (int e = 0; e < NUM_EXPERTS; ++e) acc[e] = 0.0f;

  const float* xr = x + (size_t)t * C;
  for (int c = lane; c < C; c += 32) {
    const float xv = xr[c];
    const float* wr = Wr + (size_t)c * NUM_EXPERTS;
#pragma unroll
    for (int e = 0; e < NUM_EXPERTS; ++e) acc[e] += xv * wr[e];
  }
#pragma unroll
  for (int e = 0; e < NUM_EXPERTS; ++e) {
#pragma unroll
    for (int off = 16; off >= 1; off >>= 1)
      acc[e] += __shfl_xor(acc[e], off, 32);
  }
  if (lane == 0) {
    float b0 = -1e30f, b1 = -1e30f;
    int i0 = 0, i1 = 0;
#pragma unroll
    for (int e = 0; e < NUM_EXPERTS; ++e) {   // strict '>' => lower index wins ties
      float v = acc[e];
      if (v > b0)      { b1 = b0; i1 = i0; b0 = v; i0 = e; }
      else if (v > b1) { b1 = v;  i1 = e; }
    }
    float w1 = __expf(b1 - b0);
    float inv = 1.0f / (1.0f + w1);
    float w0 = inv;
    w1 *= inv;
    int s0 = atomicAdd(&counts[i0], 1);
    tok_list[i0 * N + s0] = t; gate_list[i0 * N + s0] = w0;
    int s1 = atomicAdd(&counts[i1], 1);
    tok_list[i1 * N + s1] = t; gate_list[i1 * N + s1] = w1;
  }
}

// --------------------------------------------------- WMMA fragment load -----
// LDS tile row-major [row][32] bf16; two ds_load_b128 per fragment.
__device__ __forceinline__ v16bf frag_ld(const __bf16* lds, int row, int half) {
  const v8bf lo = *(const v8bf*)(lds + row * 32 + half * 8);
  const v8bf hi = *(const v8bf*)(lds + row * 32 + 16 + half * 8);
  return __builtin_shufflevector(lo, hi, 0, 1, 2, 3, 4, 5, 6, 7,
                                 8, 9, 10, 11, 12, 13, 14, 15);
}

__device__ __forceinline__ void mma_tile(const __bf16* aBuf, const __bf16* bBuf,
                                         int wm, int wn, int l16, int half,
                                         v8f (&acc)[2][4]) {
  v16bf a[2], b[4];
#pragma unroll
  for (int rm = 0; rm < 2; ++rm) a[rm] = frag_ld(aBuf, wm * 32 + rm * 16 + l16, half);
#pragma unroll
  for (int cn = 0; cn < 4; ++cn) b[cn] = frag_ld(bBuf, wn * 64 + cn * 16 + l16, half);
#pragma unroll
  for (int rm = 0; rm < 2; ++rm)
#pragma unroll
    for (int cn = 0; cn < 4; ++cn)
      acc[rm][cn] = __builtin_amdgcn_wmma_f32_16x16x32_bf16(
          false, a[rm], false, b[cn], (short)0, acc[rm][cn], false, false);
}

// Pack 16 f32 (4×v4f) -> two v8bf and store as two ds_store_b128.
__device__ __forceinline__ void store_row16(__bf16* dst, const v4f (&v)[4]) {
  v8bf p0, p1;
#pragma unroll
  for (int j = 0; j < 4; ++j) {
    p0[j]     = (__bf16)v[0][j];
    p0[j + 4] = (__bf16)v[1][j];
    p1[j]     = (__bf16)v[2][j];
    p1[j + 4] = (__bf16)v[3][j];
  }
  *(v8bf*)(dst)     = p0;
  *(v8bf*)(dst + 8) = p1;
}

// Store a 4k x 4h block K-transposed: one ds_store_b64 per h.
__device__ __forceinline__ void store_bT(__bf16* bBuf, int hb, int kr,
                                         const v4f (&v)[4]) {
#pragma unroll
  for (int j = 0; j < 4; ++j) {
    v4bf q;
#pragma unroll
    for (int i = 0; i < 4; ++i) q[i] = (__bf16)v[i][j];
    *(v4bf*)(bBuf + (hb + j) * 32 + kr) = q;
  }
}

// -------------------------------------------------------------- GEMM 1 ------
// h[row, hBase..] = gelu( gather(x)[row, :C] @ W1[e][:, hBase..] + b1[e] )
__global__ __launch_bounds__(256)
void moe_gemm1(const float* __restrict__ x, const float* __restrict__ W1,
               const float* __restrict__ b1, const int* __restrict__ counts,
               const int* __restrict__ tok_list, __bf16* __restrict__ hbuf,
               int e, int N, int C, int H) {
  const int cnt = counts[e];
  const int rowBase = blockIdx.y * 128;
  if (rowBase >= cnt) return;
  const int hBase = blockIdx.x * 128;
  const int* toks = tok_list + (size_t)e * N;
  const float* W1e = W1 + (size_t)e * C * H;

  __shared__ __align__(16) __bf16 aLds[2][128 * 32];  // [token][k]
  __shared__ __align__(16) __bf16 bLds[2][128 * 32];  // [h][k] (K-transposed)

  const int tid = threadIdx.x;
  const int lane = tid & 31;
  const int waveId = tid >> 5;
  const int wm = waveId & 3;
  const int wn = waveId >> 2;
  const int half = lane >> 4;
  const int l16 = lane & 15;

  // A staging: thread owns row ar, 16 k's starting at acb
  const int ar = tid >> 1, acb = (tid & 1) * 16;
  const int agrow = rowBase + ar;
  const float* aSrc = (agrow < cnt) ? (x + (size_t)toks[agrow] * C + acb) : nullptr;
  // B staging: thread owns 4 k-rows at bkr, 4 h's at bhb
  const int bkr = (tid >> 5) * 4, bhb = (tid & 31) * 4;
  const float* bSrc = W1e + (size_t)bkr * H + hBase + bhb;

  v8f acc[2][4];
#pragma unroll
  for (int rm = 0; rm < 2; ++rm)
#pragma unroll
    for (int cn = 0; cn < 4; ++cn)
#pragma unroll
      for (int i = 0; i < 8; ++i) acc[rm][cn][i] = 0.0f;

  v4f aReg[4], bReg[4];
  auto load_regs = [&](int k0) {
#pragma unroll
    for (int q = 0; q < 4; ++q) {
      if (aSrc) aReg[q] = *(const v4f*)(aSrc + k0 + q * 4);
      else      aReg[q] = (v4f)(0.0f);
      bReg[q] = *(const v4f*)(bSrc + (size_t)(k0 + q) * H);
    }
  };
  auto store_lds = [&](int buf) {
    store_row16(&aLds[buf][ar * 32 + acb], aReg);
    store_bT(&bLds[buf][0], bhb, bkr, bReg);
  };

  load_regs(0);
  store_lds(0);
  __syncthreads();
  int cur = 0;
  for (int k0 = 32; k0 < C; k0 += 32) {
    load_regs(k0);                                    // prefetch tile k0 to regs
    mma_tile(aLds[cur], bLds[cur], wm, wn, l16, half, acc);
    store_lds(cur ^ 1);
    __syncthreads();
    cur ^= 1;
  }
  mma_tile(aLds[cur], bLds[cur], wm, wn, l16, half, acc);

  // epilogue: + b1, exact GELU, bf16 store
#pragma unroll
  for (int rm = 0; rm < 2; ++rm)
#pragma unroll
    for (int cn = 0; cn < 4; ++cn) {
      const int col = hBase + wn * 64 + cn * 16 + l16;
      const float bias = b1[(size_t)e * H + col];
#pragma unroll
      for (int r = 0; r < 8; ++r) {
        const int m = wm * 32 + rm * 16 + r + 8 * half;
        const int grow = rowBase + m;
        if (grow < cnt) {
          float v = acc[rm][cn][r] + bias;
          v = 0.5f * v * (1.0f + erff(v * 0.70710678118654752f));
          hbuf[(size_t)grow * H + col] = (__bf16)v;
        }
      }
    }
}

// -------------------------------------------------------------- GEMM 2 ------
// out[tok, cBase..] += gate * ( h[row, :H] @ W2[e][:, cBase..] + b2[e] )
__global__ __launch_bounds__(256)
void moe_gemm2(const __bf16* __restrict__ hbuf, const float* __restrict__ W2,
               const float* __restrict__ b2, const int* __restrict__ counts,
               const int* __restrict__ tok_list, const float* __restrict__ gate_list,
               float* __restrict__ out, int e, int N, int C, int H) {
  const int cnt = counts[e];
  const int rowBase = blockIdx.y * 128;
  if (rowBase >= cnt) return;
  const int cBase = blockIdx.x * 128;
  const int* toks = tok_list + (size_t)e * N;
  const float* gates = gate_list + (size_t)e * N;
  const float* W2e = W2 + (size_t)e * H * C;

  __shared__ __align__(16) __bf16 aLds[2][128 * 32];
  __shared__ __align__(16) __bf16 bLds[2][128 * 32];

  const int tid = threadIdx.x;
  const int lane = tid & 31;
  const int waveId = tid >> 5;
  const int wm = waveId & 3;
  const int wn = waveId >> 2;
  const int half = lane >> 4;
  const int l16 = lane & 15;

  const int ar = tid >> 1, acb = (tid & 1) * 16;
  const int agrow = rowBase + ar;
  const __bf16* aSrc = (agrow < cnt) ? (hbuf + (size_t)agrow * H + acb) : nullptr;
  const int bkr = (tid >> 5) * 4, bcb = (tid & 31) * 4;
  const float* bSrc = W2e + (size_t)bkr * C + cBase + bcb;

  v8f acc[2][4];
#pragma unroll
  for (int rm = 0; rm < 2; ++rm)
#pragma unroll
    for (int cn = 0; cn < 4; ++cn)
#pragma unroll
      for (int i = 0; i < 8; ++i) acc[rm][cn][i] = 0.0f;

  uint4 aReg[2];
  v4f bReg[4];
  auto load_regs = [&](int k0) {
    if (aSrc) {
      aReg[0] = *(const uint4*)(aSrc + k0);
      aReg[1] = *(const uint4*)(aSrc + k0 + 8);
    } else {
      uint4 z; z.x = z.y = z.z = z.w = 0u;
      aReg[0] = z; aReg[1] = z;
    }
#pragma unroll
    for (int q = 0; q < 4; ++q)
      bReg[q] = *(const v4f*)(bSrc + (size_t)(k0 + q) * C);
  };
  auto store_lds = [&](int buf) {
    uint4* dst = (uint4*)(&aLds[buf][ar * 32 + acb]);
    dst[0] = aReg[0];
    dst[1] = aReg[1];
    store_bT(&bLds[buf][0], bcb, bkr, bReg);
  };

  load_regs(0);
  store_lds(0);
  __syncthreads();
  int cur = 0;
  for (int k0 = 32; k0 < H; k0 += 32) {
    load_regs(k0);
    mma_tile(aLds[cur], bLds[cur], wm, wn, l16, half, acc);
    store_lds(cur ^ 1);
    __syncthreads();
    cur ^= 1;
  }
  mma_tile(aLds[cur], bLds[cur], wm, wn, l16, half, acc);

  // epilogue: +b2, * gate, scatter-accumulate (token unique per expert,
  // expert launches serialize on the stream -> no atomics needed)
#pragma unroll
  for (int rm = 0; rm < 2; ++rm)
#pragma unroll
    for (int cn = 0; cn < 4; ++cn) {
      const int col = cBase + wn * 64 + cn * 16 + l16;
      const float bias = b2[(size_t)e * C + col];
#pragma unroll
      for (int r = 0; r < 8; ++r) {
        const int m = wm * 32 + rm * 16 + r + 8 * half;
        const int grow = rowBase + m;
        if (grow < cnt) {
          const int tok = toks[grow];
          const float g = gates[grow];
          float* o = out + (size_t)tok * C + col;
          *o += g * (acc[rm][cn][r] + bias);
        }
      }
    }
}

// ---------------------------------------------------------------- host ------
extern "C" void kernel_launch(void* const* d_in, const int* in_sizes, int n_in,
                              void* d_out, int out_size, void* d_ws, size_t ws_size,
                              hipStream_t stream) {
  const float* x  = (const float*)d_in[0];
  const float* Wr = (const float*)d_in[1];
  const float* W1 = (const float*)d_in[2];
  const float* b1 = (const float*)d_in[3];
  const float* W2 = (const float*)d_in[4];
  const float* b2 = (const float*)d_in[5];
  float* out = (float*)d_out;

  const int E = NUM_EXPERTS;
  const int C = in_sizes[1] / E;   // Wr is [C, E]
  const int N = in_sizes[0] / C;   // x  is [B*T, C]
  const int H = in_sizes[3] / E;   // b1 is [E, H]

  char* ws = (char*)d_ws;
  int* counts = (int*)ws;                                   // E ints
  size_t off = 256;
  int* tok_list = (int*)(ws + off);      off += (size_t)E * N * sizeof(int);
  float* gate_list = (float*)(ws + off); off += (size_t)E * N * sizeof(float);
  off = (off + 255) & ~(size_t)255;
  __bf16* hbuf = (__bf16*)(ws + off);                        // N*H bf16

  moe_init<<<(out_size + 255) / 256, 256, 0, stream>>>(out, out_size, counts);
  moe_route<<<(N + 7) / 8, 256, 0, stream>>>(x, Wr, counts, tok_list, gate_list, N, C);

  dim3 g1(H / 128, (N + 127) / 128);
  dim3 g2(C / 128, (N + 127) / 128);
  for (int e = 0; e < NUM_EXPERTS; ++e) {
    moe_gemm1<<<g1, 256, 0, stream>>>(x, W1, b1, counts, tok_list, hbuf, e, N, C, H);
    moe_gemm2<<<g2, 256, 0, stream>>>(hbuf, W2, b2, counts, tok_list, gate_list, out,
                                      e, N, C, H);
  }
}
